// QLSTM_65481071397319
// MI455X (gfx1250) — matrix operand order repaired and checked by reading, not compiled, not run on hardware
//
#include <hip/hip_runtime.h>
#include <hip/hip_bf16.h>

// ---------------------------------------------------------------------------
// 2-layer LSTM for MI455X (gfx1250).
// Persistent-grid recurrence, bf16 WMMA (v_wmma_f32_16x16x32_bf16, f32 acc).
// Software-pipelined: phase p computes layer0(p) AND layer1(p-1) -> a single
// grid barrier per timestep (513 total instead of 1024).
// Weights tile-packed to bf16 B-fragments once per launch (L2-resident, 28MB).
// T=512, B=64, D=512, H=1024.
// ---------------------------------------------------------------------------

typedef __bf16 v16bf __attribute__((ext_vector_type(16)));
typedef __bf16 v8bf  __attribute__((ext_vector_type(8)));
typedef float  v8f   __attribute__((ext_vector_type(8)));
typedef float  v4f   __attribute__((ext_vector_type(4)));

constexpr int TT = 512;     // timesteps
constexpr int BB = 64;      // batch
constexpr int DD = 512;     // input dim
constexpr int HH = 1024;    // hidden dim
constexpr int NWG = 64;     // workgroups = H/16 column groups
constexpr int STATE = BB * HH;                       // 65536 elements
constexpr size_t OUT_STATES = (size_t)TT * BB * HH;  // final-states offset in d_out

// ---------------------------------------------------------------------------
// Activations (lower to v_exp_f32)
// ---------------------------------------------------------------------------
__device__ __forceinline__ float fsigmoid(float v) {
    return 1.0f / (1.0f + __expf(-v));
}
__device__ __forceinline__ float ftanh_(float v) {
    return 2.0f / (1.0f + __expf(-2.0f * v)) - 1.0f;
}

// ---------------------------------------------------------------------------
// WMMA fragment loaders (this code's contiguous k order, same for A and B).
// A (16x32 bf16): lane<16: row=lane,    K[0..15]
//                 lane>=16: row=lane-16, K[16..31]
// B (32x16 bf16): lane<16: col=lane, K[0..15]; lane>=16: col=lane-16, K[16..31]
// B pre-packed so each lane's 16 halfs are contiguous (one 32B vector load).
// ---------------------------------------------------------------------------
__device__ __forceinline__ v16bf load_a_frag_bf(const __bf16* __restrict__ A,
                                                int rowStride, int M, int kt, int lane) {
    const int klo = (lane & 16) ? 16 : 0;
    const __bf16* p = A + (size_t)M * rowStride + kt * 32 + klo;
    union { v8bf h[2]; v16bf v; } u;
    u.h[0] = *(const v8bf*)(p);
    u.h[1] = *(const v8bf*)(p + 8);
    return u.v;
}

__device__ __forceinline__ v16bf load_b_frag(const __bf16* __restrict__ Wp,
                                             int ktiles, int wg, int g, int kt, int lane) {
    return *(const v16bf*)(Wp + (((size_t)((g * 64 + wg) * ktiles + kt)) << 9) + (lane << 4));
}

// acc[g] += A x Wp^T for the 4 gate column tiles of workgroup `wg`.
__device__ __forceinline__ void gemm_bf16_acc(const __bf16* __restrict__ Aa, int Ka,
                                              const __bf16* __restrict__ Wp, int ktiles,
                                              int wg, int M, int lane, v8f (&acc)[4]) {
    for (int kt = 0; kt < ktiles; ++kt) {
        v16bf a = load_a_frag_bf(Aa, Ka, M, kt, lane);
#pragma unroll
        for (int g = 0; g < 4; ++g) {
            v16bf b = load_b_frag(Wp, ktiles, wg, g, kt, lane);
            acc[g] = __builtin_amdgcn_wmma_f32_16x16x32_bf16(
                false, a, false, b, (short)0, acc[g], false, false);
        }
    }
}

// Fused: same A feeds two weight matrices (h0 -> W_hh0 and W_ih1).
// Loads each A fragment once, issues 8 independent WMMAs against it.
__device__ __forceinline__ void gemm_bf16_acc2(const __bf16* __restrict__ Aa, int Ka,
                                               const __bf16* __restrict__ WpA,
                                               const __bf16* __restrict__ WpB,
                                               int ktiles, int wg, int M, int lane,
                                               v8f (&accA)[4], v8f (&accB)[4]) {
    for (int kt = 0; kt < ktiles; ++kt) {
        v16bf a = load_a_frag_bf(Aa, Ka, M, kt, lane);
#pragma unroll
        for (int g = 0; g < 4; ++g) {
            v16bf b = load_b_frag(WpA, ktiles, wg, g, kt, lane);
            accA[g] = __builtin_amdgcn_wmma_f32_16x16x32_bf16(
                false, a, false, b, (short)0, accA[g], false, false);
        }
#pragma unroll
        for (int g = 0; g < 4; ++g) {
            v16bf b = load_b_frag(WpB, ktiles, wg, g, kt, lane);
            accB[g] = __builtin_amdgcn_wmma_f32_16x16x32_bf16(
                false, a, false, b, (short)0, accB[g], false, false);
        }
    }
}

// ---------------------------------------------------------------------------
// Grid-wide barrier (64 persistent workgroups, generation counter)
// ---------------------------------------------------------------------------
__device__ __forceinline__ void grid_sync(unsigned* cnt, unsigned* gen, int nwg) {
    __syncthreads();
    if (threadIdx.x == 0) {
        __threadfence();
        unsigned g = __atomic_load_n(gen, __ATOMIC_RELAXED);
        unsigned prev = atomicAdd(cnt, 1u);
        if (prev == (unsigned)(nwg - 1)) {
            __atomic_store_n(cnt, 0u, __ATOMIC_RELAXED);
            __threadfence();
            atomicAdd(gen, 1u);
        } else {
            while (__atomic_load_n(gen, __ATOMIC_RELAXED) == g) {
                __builtin_amdgcn_s_sleep(2);
            }
        }
        __threadfence();
    }
    __syncthreads();
}

// ---------------------------------------------------------------------------
// Prep kernels
// ---------------------------------------------------------------------------
// Pack W [4096 x K] fp32 into bf16 B-fragment tiles of W^T.
__global__ void pack_w(const float* __restrict__ W, __bf16* __restrict__ dst, int K) {
    int tid = blockIdx.x * blockDim.x + threadIdx.x;
    int ktiles = K >> 5;
    int total = 256 * ktiles * 32;
    if (tid >= total) return;
    int lane = tid & 31;
    int tile = tid >> 5;
    int kt = tile % ktiles;
    int nt = tile / ktiles;
    int n  = nt * 16 + (lane & 15);
    int kb = kt * 32 + ((lane >> 4) << 4);
    const float* src = W + (size_t)n * K + kb;
    __bf16* d = dst + ((size_t)tile << 9) + (lane << 4);
#pragma unroll
    for (int j = 0; j < 16; ++j) d[j] = (__bf16)src[j];
}

// Convert inputs [T*B*D] fp32 -> bf16 (8 elements / thread, vectorized).
__global__ void cvt_x(const float* __restrict__ src, __bf16* __restrict__ dst, int n8) {
    int i = blockIdx.x * blockDim.x + threadIdx.x;
    if (i >= n8) return;
    const float* s = src + (size_t)i * 8;
    v4f a = *(const v4f*)(s);
    v4f b = *(const v4f*)(s + 4);
    v8bf o;
#pragma unroll
    for (int j = 0; j < 4; ++j) {
        o[j]     = (__bf16)a[j];
        o[4 + j] = (__bf16)b[j];
    }
    *(v8bf*)(dst + (size_t)i * 8) = o;
}

__global__ void prep_bias(const float* __restrict__ bi, const float* __restrict__ bh,
                          float* __restrict__ dst) {
    int i = blockIdx.x * blockDim.x + threadIdx.x;
    if (i < 4 * HH) dst[i] = bi[i] + bh[i];
}

__global__ void init_state(__bf16* __restrict__ h0, __bf16* __restrict__ h1,
                           float* __restrict__ c0, float* __restrict__ c1,
                           unsigned* __restrict__ sync) {
    int i = blockIdx.x * blockDim.x + threadIdx.x;
    if (i < 2 * STATE) { h0[i] = (__bf16)0.0f; h1[i] = (__bf16)0.0f; }
    if (i < STATE)     { c0[i] = 0.0f;         c1[i] = 0.0f; }
    if (i < 2)         { sync[i * 64] = 0u; }
}

// ---------------------------------------------------------------------------
// Persistent pipelined recurrence: 64 WGs x 128 threads (4 wave32 each).
// WG wg owns hidden columns [16*wg,16*wg+16); wave w owns batch rows
// [16*w,16*w+16). Phase p (p = 0..T):
//   layer0(p)   : h0(p)   = cell(x(p)   @Wih0 + h0(p-1)@Whh0)   [p <  T]
//   layer1(p-1) : h1(p-1) = cell(h0(p-1)@Wih1 + h1(p-2)@Whh1)   [p >= 1]
// Both consume only data sealed by the barrier at the end of phase p-1, so
// ONE grid barrier per phase suffices.
// Buffering: h0(q) lives in bufH0[q&1]; h1(q) lives in bufH1[q&1... see
// pointers below]; c0/c1 are lane-private (no buffering needed).
// ---------------------------------------------------------------------------
__global__ void __launch_bounds__(128, 1)
lstm_persistent(const __bf16* __restrict__ xbf,
                const __bf16* __restrict__ Wp_ih0,
                const __bf16* __restrict__ Wp_hh0,
                const __bf16* __restrict__ Wp_ih1,
                const __bf16* __restrict__ Wp_hh1,
                const float* __restrict__ bias0,
                const float* __restrict__ bias1,
                __bf16* __restrict__ h0bf,   // 2 x [B*H]
                __bf16* __restrict__ h1bf,   // 2 x [B*H]
                float* __restrict__ c0,
                float* __restrict__ c1,
                float* __restrict__ out,
                unsigned* __restrict__ sync) {
    const int wg   = blockIdx.x;
    const int wave = threadIdx.x >> 5;
    const int lane = threadIdx.x & 31;
    const int M    = (wave << 4) + (lane & 15);           // A-fragment row
    const int nCol = (wg << 4) + (lane & 15);             // owned hidden column
    const int rowBase = (wave << 4) + ((lane >> 4) << 3); // D-fragment row base

    unsigned* cnt = sync;
    unsigned* gen = sync + 64;

    // Per-lane gate biases (one per gate), loaded once.
    float b0[4], b1[4];
#pragma unroll
    for (int g = 0; g < 4; ++g) {
        b0[g] = bias0[(g << 10) + nCol];
        b1[g] = bias1[(g << 10) + nCol];
    }

    for (int p = 0; p <= TT; ++p) {
        // h0(p-1) was written to bufH0[(p-1)&1]; h0(p) goes to bufH0[p&1].
        const __bf16* h0_prev = h0bf + ((p + 1) & 1) * STATE;  // h0(p-1)
        __bf16*       h0_cur  = h0bf + (p & 1) * STATE;        // h0(p)
        // h1(q) lives in bufH1[q&1]: read h1(p-2), write h1(p-1).
        const __bf16* h1_prev = h1bf + (p & 1) * STATE;        // h1(p-2)
        __bf16*       h1_cur  = h1bf + ((p + 1) & 1) * STATE;  // h1(p-1)

        v8f acc0[4], acc1[4];
        const bool doL0 = (p < TT);
        const bool doL1 = (p >= 1);

        if (doL0) {
#pragma unroll
            for (int g = 0; g < 4; ++g)
#pragma unroll
                for (int r = 0; r < 8; ++r) acc0[g][r] = b0[g];
            // x(p) @ W_ih0^T  (K = 512)
            gemm_bf16_acc(xbf + (size_t)p * BB * DD, DD, Wp_ih0, DD >> 5,
                          wg, M, lane, acc0);
        }
        if (doL1) {
#pragma unroll
            for (int g = 0; g < 4; ++g)
#pragma unroll
                for (int r = 0; r < 8; ++r) acc1[g][r] = b1[g];
        }

        if (doL0 && doL1) {
            // h0(p-1) feeds BOTH W_hh0 (layer0) and W_ih1 (layer1):
            // load each A fragment once, 8 WMMAs per fragment.
            gemm_bf16_acc2(h0_prev, HH, Wp_hh0, Wp_ih1, HH >> 5,
                           wg, M, lane, acc0, acc1);
            gemm_bf16_acc(h1_prev, HH, Wp_hh1, HH >> 5, wg, M, lane, acc1);
        } else if (doL0) {          // p == 0
            gemm_bf16_acc(h0_prev, HH, Wp_hh0, HH >> 5, wg, M, lane, acc0);
        } else {                    // p == TT
            gemm_bf16_acc(h0_prev, HH, Wp_ih1, HH >> 5, wg, M, lane, acc1);
            gemm_bf16_acc(h1_prev, HH, Wp_hh1, HH >> 5, wg, M, lane, acc1);
        }

        // ---------------- layer0 cell update: h0(p), c0 ----------------
        if (doL0) {
#pragma unroll
            for (int r = 0; r < 8; ++r) {
                const size_t idx = (size_t)(rowBase + r) * HH + nCol;
                float iv = fsigmoid(acc0[0][r]);
                float fv = fsigmoid(acc0[1][r]);
                float gv = ftanh_(acc0[2][r]);
                float ov = fsigmoid(acc0[3][r]);
                float cn = fv * c0[idx] + iv * gv;
                float hn = ov * ftanh_(cn);
                c0[idx] = cn;
                h0_cur[idx] = (__bf16)hn;
                if (p == TT - 1) {
                    out[OUT_STATES + idx] = hn;                     // final h0
                    out[OUT_STATES + (size_t)STATE + idx] = cn;     // final c0
                }
            }
        }
        // ---------------- layer1 cell update: h1(p-1), c1, output -------
        if (doL1) {
            const int t = p - 1;
#pragma unroll
            for (int r = 0; r < 8; ++r) {
                const size_t idx = (size_t)(rowBase + r) * HH + nCol;
                float iv = fsigmoid(acc1[0][r]);
                float fv = fsigmoid(acc1[1][r]);
                float gv = ftanh_(acc1[2][r]);
                float ov = fsigmoid(acc1[3][r]);
                float cn = fv * c1[idx] + iv * gv;
                float hn = ov * ftanh_(cn);
                c1[idx] = cn;
                h1_cur[idx] = (__bf16)hn;
                out[(size_t)t * STATE + idx] = hn;                  // sequence out
                if (p == TT) {
                    out[OUT_STATES + 2 * (size_t)STATE + idx] = hn; // final h1
                    out[OUT_STATES + 3 * (size_t)STATE + idx] = cn; // final c1
                }
            }
        }
        grid_sync(cnt, gen, NWG);
    }
}

// ---------------------------------------------------------------------------
// Launch
// ---------------------------------------------------------------------------
extern "C" void kernel_launch(void* const* d_in, const int* in_sizes, int n_in,
                              void* d_out, int out_size, void* d_ws, size_t ws_size,
                              hipStream_t stream) {
    const float* x     = (const float*)d_in[0];
    const float* W_ih0 = (const float*)d_in[1];
    const float* W_hh0 = (const float*)d_in[2];
    const float* b_ih0 = (const float*)d_in[3];
    const float* b_hh0 = (const float*)d_in[4];
    const float* W_ih1 = (const float*)d_in[5];
    const float* W_hh1 = (const float*)d_in[6];
    const float* b_ih1 = (const float*)d_in[7];
    const float* b_hh1 = (const float*)d_in[8];
    float* out = (float*)d_out;

    char* ws = (char*)d_ws;
    size_t off = 0;
    auto alloc = [&](size_t bytes) -> void* {
        off = (off + 255) & ~(size_t)255;
        void* p = ws + off;
        off += bytes;
        return p;
    };

    __bf16* Wp_ih0 = (__bf16*)alloc((size_t)4096 * 512 * 2);
    __bf16* Wp_hh0 = (__bf16*)alloc((size_t)4096 * 1024 * 2);
    __bf16* Wp_ih1 = (__bf16*)alloc((size_t)4096 * 1024 * 2);
    __bf16* Wp_hh1 = (__bf16*)alloc((size_t)4096 * 1024 * 2);
    __bf16* xbf    = (__bf16*)alloc((size_t)TT * BB * DD * 2);
    float*  bias0  = (float*)alloc((size_t)4096 * 4);
    float*  bias1  = (float*)alloc((size_t)4096 * 4);
    __bf16* h0bf   = (__bf16*)alloc((size_t)2 * STATE * 2);
    __bf16* h1bf   = (__bf16*)alloc((size_t)2 * STATE * 2);
    float*  c0     = (float*)alloc((size_t)STATE * 4);
    float*  c1     = (float*)alloc((size_t)STATE * 4);
    unsigned* syncp = (unsigned*)alloc(1024);

    // Weight packing: 256 N-tiles x (K/32) K-tiles x 32 lanes threads
    {
        int thr0 = 256 * (512 / 32) * 32;    // 131072
        int thr1 = 256 * (1024 / 32) * 32;   // 262144
        pack_w<<<(thr0 + 255) / 256, 256, 0, stream>>>(W_ih0, Wp_ih0, 512);
        pack_w<<<(thr1 + 255) / 256, 256, 0, stream>>>(W_hh0, Wp_hh0, 1024);
        pack_w<<<(thr1 + 255) / 256, 256, 0, stream>>>(W_ih1, Wp_ih1, 1024);
        pack_w<<<(thr1 + 255) / 256, 256, 0, stream>>>(W_hh1, Wp_hh1, 1024);
    }
    {
        int n8 = TT * BB * DD / 8;           // 2097152
        cvt_x<<<(n8 + 255) / 256, 256, 0, stream>>>(x, xbf, n8);
    }
    prep_bias<<<(4 * HH + 255) / 256, 256, 0, stream>>>(b_ih0, b_hh0, bias0);
    prep_bias<<<(4 * HH + 255) / 256, 256, 0, stream>>>(b_ih1, b_hh1, bias1);
    init_state<<<(2 * STATE + 255) / 256, 256, 0, stream>>>(h0bf, h1bf, c0, c1, syncp);

    lstm_persistent<<<NWG, 128, 0, stream>>>(xbf, Wp_ih0, Wp_hh0, Wp_ih1, Wp_hh1,
                                             bias0, bias1, h0bf, h1bf, c0, c1,
                                             out, syncp);
}
